// RNN_61237643706497
// MI455X (gfx1250) — matrix operand order repaired and checked by reading, not compiled, not run on hardware
//
#include <hip/hip_runtime.h>

typedef __attribute__((ext_vector_type(16))) _Float16 v16h;
typedef __attribute__((ext_vector_type(8)))  float    v8f;

#define ALPHA_ 0.2f
#define B_ 64
#define T_ 512
#define I_ 128
#define H_ 512
#define O_ 64

// A-operand (16x32 f16) per-lane K mapping: half index e (0..15), hi = lane/16
// VGPR v<4: K = 2v+p + 8*hi ; v>=4: K = 16 + 2(v-4)+p + 8*hi  =>  K = (e&7) + 8*hi + ((e&8)<<1)
__device__ __forceinline__ int amapK(int e, int hi) {
    return (e & 7) + 8 * hi + ((e & 8) << 1);
}

// Pack W[n][k] (row-major, leading dim ldk) into WMMA B-operand f16 layout:
// dst[((ntile*nkb + kb)*32 + lane)*16 + e] = W[ntile*16 + lane%16][kb*32 + amapK(e, lane/16)]
// (B 32x16 f16: lane holds column n = lane%16; halves e hold K = amapK(e, lane/16))
__global__ void pack_b_kernel(const float* __restrict__ W, _Float16* __restrict__ dst,
                              int ntiles, int nkb, int ldk) {
    int total = ntiles * nkb * 512;
    for (int idx = blockIdx.x * blockDim.x + threadIdx.x; idx < total;
         idx += gridDim.x * blockDim.x) {
        int e    = idx & 15;
        int lane = (idx >> 4) & 31;
        int kb   = (idx >> 9) % nkb;
        int nt   = idx / (nkb * 512);
        int n    = nt * 16 + (lane & 15);
        int k    = kb * 32 + amapK(e, lane >> 4);
        dst[idx] = (_Float16)W[(size_t)n * ldk + k];
    }
}

// Phase 1: u[t][b][h] = sum_i x[b][t][i] * W_ih[h][i]   (einsum 'bti,hi->tbh')
// grid = (4 batch tiles, 512 time steps), 512 threads = 16 waves, 2 N-tiles per wave.
__global__ __launch_bounds__(512) void uproj_kernel(const float* __restrict__ x,
                                                    const _Float16* __restrict__ wih_pk,
                                                    float* __restrict__ u) {
    __shared__ __align__(32) _Float16 xA[4 * 32 * 16]; // A layout, 4 K-blocks of 32
    const int bt = blockIdx.x, t = blockIdx.y, tid = threadIdx.x;
    const int b0 = bt * 16;
    for (int s = 0; s < 4; ++s) {
        int idx  = s * 512 + tid;
        int e    = idx & 15;
        int lane = (idx >> 4) & 31;
        int kb   = idx >> 9;
        int m    = lane & 15;
        int k    = kb * 32 + amapK(e, lane >> 4);
        xA[idx] = (_Float16)x[((size_t)(b0 + m) * T_ + t) * I_ + k];
    }
    __syncthreads();
    const int w = tid >> 5, lane = tid & 31;
    const int hi_c = lane >> 4, N = lane & 15;
    for (int sub = 0; sub < 2; ++sub) {
        int ntile = w * 2 + sub;
        v8f acc = {};
        for (int kb = 0; kb < 4; ++kb) {
            v16h a = *(const v16h*)(&xA[(kb * 32 + lane) * 16]);
            v16h b = *(const v16h*)(&wih_pk[(((size_t)ntile * 4 + kb) * 32 + lane) * 16]);
            acc = __builtin_amdgcn_wmma_f32_16x16x32_f16(false, a, false, b,
                                                         (short)0, acc, false, false);
        }
        int n = ntile * 16 + N;
        for (int g = 0; g < 8; ++g) {
            int m = g + 8 * hi_c;
            u[((size_t)t * B_ + b0 + m) * H_ + n] = acc[g];
        }
    }
}

// Phase 2: persistent recurrence. 4 independent workgroups (one per 16 batch rows),
// 512 threads = 16 waves. h-state in registers, r in LDS pre-packed as WMMA A operand.
__global__ __launch_bounds__(512) void rnn_step_kernel(
    const float* __restrict__ u,          // [T][B][H] f32
    const _Float16* __restrict__ whh_pk,  // [32 ntile][16 kb][32 lane][16 e]
    const _Float16* __restrict__ wout_pk, // [4  ntile][16 kb][32 lane][16 e]
    const float* __restrict__ bias,       // [H]
    const float* __restrict__ h0,         // [H]
    const float* __restrict__ b_out,      // [O]
    float* __restrict__ os, float* __restrict__ hs, float* __restrict__ rs) {
    __shared__ __align__(32) _Float16 rA[16 * 32 * 16]; // 16 KB: [kb][lane][e]
    const int b0 = blockIdx.x * 16;
    const int tid = threadIdx.x;
    const int w = tid >> 5, lane = tid & 31;
    const int hi_c = lane >> 4, N = lane & 15;

    // r_init = relu(broadcast h0) in A layout
    for (int s = 0; s < 16; ++s) {
        int idx  = s * 512 + tid;
        int e    = idx & 15;
        int la   = (idx >> 4) & 31;
        int kb   = idx >> 9;
        int k    = kb * 32 + amapK(e, la >> 4);
        float v  = h0[k];
        rA[idx] = (_Float16)(v > 0.f ? v : 0.f);
    }

    const int nt0 = 2 * w, nt1 = 2 * w + 1;
    const int n0 = nt0 * 16 + N, n1 = nt1 * 16 + N;
    const float bias0 = bias[n0], bias1 = bias[n1];
    v8f hv0, hv1;
    for (int g = 0; g < 8; ++g) { hv0[g] = h0[n0]; hv1[g] = h0[n1]; }

    // LDS scatter slots for writing new r (A layout), precomputed (n fixed per thread)
    auto slotbase = [&](int n) -> int {
        int kb2 = n >> 5, kk = n & 31;
        int hia = (kk >> 3) & 1;
        int e   = (kk & 7) + ((kk & 16) >> 1);
        return (kb2 * 32 + 8 * hi_c + 16 * hia) * 16 + e;
    };
    const int sb0 = slotbase(n0), sb1 = slotbase(n1);

    const int on = w * 16 + N;                       // O-proj column (waves 0..3)
    const float bout = (w < 4) ? b_out[on] : 0.f;

    __syncthreads();

    for (int t = 0; t < T_; ++t) {
        if (t + 1 < T_) { // prefetch next step's u rows into cache hierarchy
            const float* un = u + (size_t)(t + 1) * B_ * H_;
            __builtin_prefetch(&un[(size_t)(b0 + 8 * hi_c) * H_ + n0], 0, 1);
        }
        v8f acc0 = {}, acc1 = {};
        for (int kb = 0; kb < 16; ++kb) {
            v16h a   = *(const v16h*)(&rA[(kb * 32 + lane) * 16]);
            v16h bb0 = *(const v16h*)(&whh_pk[(((size_t)nt0 * 16 + kb) * 32 + lane) * 16]);
            v16h bb1 = *(const v16h*)(&whh_pk[(((size_t)nt1 * 16 + kb) * 32 + lane) * 16]);
            acc0 = __builtin_amdgcn_wmma_f32_16x16x32_f16(false, a, false, bb0,
                                                          (short)0, acc0, false, false);
            acc1 = __builtin_amdgcn_wmma_f32_16x16x32_f16(false, a, false, bb1,
                                                          (short)0, acc1, false, false);
        }
        __syncthreads(); // all waves done reading old r (and prior-step O-proj reads)
        const float* urow = u + (size_t)t * B_ * H_;
        for (int g = 0; g < 8; ++g) {
            int m = g + 8 * hi_c;
            int b = b0 + m;
            float u0 = urow[(size_t)b * H_ + n0];
            float u1 = urow[(size_t)b * H_ + n1];
            float hn0 = (1.f - ALPHA_) * hv0[g] + ALPHA_ * (u0 + acc0[g] + bias0);
            float hn1 = (1.f - ALPHA_) * hv1[g] + ALPHA_ * (u1 + acc1[g] + bias1);
            hv0[g] = hn0; hv1[g] = hn1;
            float r0 = hn0 > 0.f ? hn0 : 0.f;
            float r1 = hn1 > 0.f ? hn1 : 0.f;
            size_t rowbase = ((size_t)b * T_ + t) * H_;
            hs[rowbase + n0] = hn0; hs[rowbase + n1] = hn1;
            rs[rowbase + n0] = r0;  rs[rowbase + n1] = r1;
            rA[sb0 + g * 16] = (_Float16)r0;
            rA[sb1 + g * 16] = (_Float16)r1;
        }
        __syncthreads(); // new r fully in LDS
        if (w < 4) {     // wave-uniform branch: EXEC all-ones inside (WMMA-legal)
            v8f acco = {};
            for (int kb = 0; kb < 16; ++kb) {
                v16h a  = *(const v16h*)(&rA[(kb * 32 + lane) * 16]);
                v16h bo = *(const v16h*)(&wout_pk[(((size_t)w * 16 + kb) * 32 + lane) * 16]);
                acco = __builtin_amdgcn_wmma_f32_16x16x32_f16(false, a, false, bo,
                                                              (short)0, acco, false, false);
            }
            for (int g = 0; g < 8; ++g) {
                int b = b0 + g + 8 * hi_c;
                os[((size_t)b * T_ + t) * O_ + on] = acco[g] + bout;
            }
        }
    }
}

extern "C" void kernel_launch(void* const* d_in, const int* in_sizes, int n_in,
                              void* d_out, int out_size, void* d_ws, size_t ws_size,
                              hipStream_t stream) {
    const float* x     = (const float*)d_in[0]; // [B,T,I]
    const float* wih   = (const float*)d_in[1]; // [H,I]
    const float* whh   = (const float*)d_in[2]; // [H,H]
    const float* bias  = (const float*)d_in[3]; // [1,H]
    const float* h0    = (const float*)d_in[4]; // [1,H]
    const float* wout  = (const float*)d_in[5]; // [O,H]
    const float* b_out = (const float*)d_in[6]; // [O]

    float* os = (float*)d_out;                        // [B,T,O]
    float* hs = os + (size_t)B_ * T_ * O_;            // [B,T,H]
    float* rs = hs + (size_t)B_ * T_ * H_;            // [B,T,H]

    char* ws = (char*)d_ws;
    float*    u       = (float*)ws;                                   // 64 MB: [T][B][H] f32
    _Float16* whh_pk  = (_Float16*)(ws + 67108864);                   // 512 KB
    _Float16* wih_pk  = (_Float16*)(ws + 67108864 + 524288);          // 128 KB
    _Float16* wout_pk = (_Float16*)(ws + 67108864 + 524288 + 131072); // 64 KB

    pack_b_kernel<<<512, 256, 0, stream>>>(whh,  whh_pk, 32, 16, 512);
    pack_b_kernel<<<128, 256, 0, stream>>>(wih,  wih_pk, 32, 4, 128);
    pack_b_kernel<<<64,  256, 0, stream>>>(wout, wout_pk, 4, 16, 512);

    dim3 g1(4, 512);
    uproj_kernel<<<g1, 512, 0, stream>>>(x, wih_pk, u);

    rnn_step_kernel<<<4, 512, 0, stream>>>(u, whh_pk, wout_pk, bias, h0, b_out,
                                           os, hs, rs);
}
